// GraphAttentionLayer_21569325761080
// MI455X (gfx1250) — matrix operand order repaired and checked by reading, not compiled, not run on hardware
//
#include <hip/hip_runtime.h>

// GAT layer for MI455X (gfx1250, wave32).
// Dominant cost: streaming 256MB adjacency once (~11us @ 23.3TB/s) while the
// attention @ Wh 17.2-GFLOP matmul runs on v_wmma_f32_16x16x32_f16 with a
// flash-attention style online softmax (one pass over adjacency).
// j-tile = 64 (two WMMA K-steps), double-buffered P tile -> 1 barrier/tile,
// global_prefetch on the adjacency stream.

#define N_NODES 8192
#define IN_DIM  256
#define OUT_DIM 128
#define ALPHA   0.2f
#define MASK_VAL (-9000000000000000.0f)

typedef __attribute__((ext_vector_type(16))) _Float16 v16h;
typedef __attribute__((ext_vector_type(8)))  _Float16 v8h;
typedef __attribute__((ext_vector_type(8)))  float    v8f;

union V16 { v16h v; v8h h[2]; };

// ---------------------------------------------------------------------------
// Kernel 0: WT[c][k] = (f16) W[k][c]   (64KB, trivial)
// ---------------------------------------------------------------------------
__global__ void gat_wt_kernel(const float* __restrict__ W, _Float16* __restrict__ WT) {
  int id = blockIdx.x * blockDim.x + threadIdx.x;
  if (id < IN_DIM * OUT_DIM) {
    int k = id >> 7;    // IN_DIM index
    int c = id & 127;   // OUT_DIM index
    WT[c * IN_DIM + k] = (_Float16)W[id];
  }
}

// ---------------------------------------------------------------------------
// Kernel 1: Wh = X @ W via WMMA; emit WhhT (f16, [OUT_DIM][N]) and
//           s1 = Wh@a1, s2 = Wh@a2 (f32).
// One block = 16 rows of Wh, 128 threads = 4 waves, 2 N-tiles per wave.
// ---------------------------------------------------------------------------
__global__ __launch_bounds__(128)
void gat_wh_kernel(const float* __restrict__ X, const _Float16* __restrict__ WT,
                   const float* __restrict__ a,
                   _Float16* __restrict__ WhhT,
                   float* __restrict__ s1, float* __restrict__ s2) {
  __shared__ _Float16 Xh[16][IN_DIM];   // 8 KB, f16 copy of X tile
  __shared__ float    Whl[16][OUT_DIM]; // 8 KB, f32 Wh tile

  const int t  = threadIdx.x;
  const int i0 = blockIdx.x * 16;

  for (int idx = t; idx < 16 * IN_DIM; idx += 128) {
    int r = idx >> 8, k = idx & 255;
    Xh[r][k] = (_Float16)X[(size_t)(i0 + r) * IN_DIM + k];
  }
  __syncthreads();

  const int w = t >> 5, l = t & 31;
  const int m = l & 15, g = l >> 4;

  v8f acc0 = {}; v8f acc1 = {};
  for (int kk = 0; kk < IN_DIM; kk += 32) {
    V16 A;
    A.h[0] = *(const v8h*)&Xh[m][kk + 16 * g];
    A.h[1] = *(const v8h*)&Xh[m][kk + 16 * g + 8];
    {
      const _Float16* bp = WT + (size_t)((2 * w + 0) * 16 + m) * IN_DIM + kk + 16 * g;
      V16 B; B.h[0] = *(const v8h*)bp; B.h[1] = *(const v8h*)(bp + 8);
      acc0 = __builtin_amdgcn_wmma_f32_16x16x32_f16(false, A.v, false, B.v,
                                                    (short)0, acc0, false, false);
    }
    {
      const _Float16* bp = WT + (size_t)((2 * w + 1) * 16 + m) * IN_DIM + kk + 16 * g;
      V16 B; B.h[0] = *(const v8h*)bp; B.h[1] = *(const v8h*)(bp + 8);
      acc1 = __builtin_amdgcn_wmma_f32_16x16x32_f16(false, A.v, false, B.v,
                                                    (short)0, acc1, false, false);
    }
  }

#pragma unroll
  for (int k = 0; k < 8; ++k) {
    Whl[k + 8 * g][(2 * w + 0) * 16 + m] = acc0[k];
    Whl[k + 8 * g][(2 * w + 1) * 16 + m] = acc1[k];
  }
  __syncthreads();

  if (t < 16) {
    float ss1 = 0.f, ss2 = 0.f;
    for (int c = 0; c < OUT_DIM; ++c) {
      float wh = Whl[t][c];
      ss1 += wh * a[c];
      ss2 += wh * a[OUT_DIM + c];
    }
    s1[i0 + t] = ss1;
    s2[i0 + t] = ss2;
  }

  {
    V16 pk;
#pragma unroll
    for (int r = 0; r < 16; ++r) pk.v[r] = (_Float16)Whl[r][t];
    _Float16* dst = WhhT + (size_t)t * N_NODES + i0;
    *(v8h*)dst       = pk.h[0];
    *(v8h*)(dst + 8) = pk.h[1];
  }
}

// ---------------------------------------------------------------------------
// Kernel 2: fused masked-softmax-attention @ Wh with online softmax.
// One block = 16 output rows, 128 threads = 4 waves, j tiled by 64
// (2 WMMA K-steps). Double-buffered P tile -> single barrier per tile.
// ---------------------------------------------------------------------------
__global__ __launch_bounds__(128)
void gat_attn_kernel(const int* __restrict__ adj, const _Float16* __restrict__ WhhT,
                     const float* __restrict__ s1, const float* __restrict__ s2,
                     float* __restrict__ out) {
  __shared__ _Float16 P[2][16][64];  // double-buffered probability tile (4 KB)
  __shared__ float    rowf[2][16];   // per-row broadcast (rescale / 1/l)

  const int t   = threadIdx.x;
  const int i0  = blockIdx.x * 16;
  const int row = t >> 3;     // 16 rows x 8 threads
  const int cg  = t & 7;      // 8 columns per thread
  const int w = t >> 5, l = t & 31;
  const int m = l & 15, g = l >> 4;

  const float s1v = s1[i0 + row];
  float m_run = -__builtin_inff();
  float l_run = 0.f;
  v8f acc0 = {}; v8f acc1 = {};

  const int* adjrow = adj + (size_t)(i0 + row) * N_NODES;

  int it = 0;
  for (int j0 = 0; j0 < N_NODES; j0 += 64, ++it) {
    const int b = it & 1;
    const int c0 = cg * 8;

    // ---- adjacency stream: 32B contiguous per thread ----
    int4   a4a = *(const int4*)(adjrow + j0 + c0);
    int4   a4b = *(const int4*)(adjrow + j0 + c0 + 4);
    float4 s2a = *(const float4*)(s2 + j0 + c0);
    float4 s2b = *(const float4*)(s2 + j0 + c0 + 4);
    if (j0 + 128 < N_NODES)  // keep the HBM pipe primed (global_prefetch_b8)
      __builtin_prefetch(adjrow + j0 + 128 + c0, 0, 1);

    int   ad[8] = {a4a.x, a4a.y, a4a.z, a4a.w, a4b.x, a4b.y, a4b.z, a4b.w};
    float sv[8] = {s2a.x, s2a.y, s2a.z, s2a.w, s2b.x, s2b.y, s2b.z, s2b.w};
    float ev[8];
    float tm = -__builtin_inff();
#pragma unroll
    for (int i = 0; i < 8; ++i) {
      float e = s1v + sv[i];
      e = e > 0.f ? e : ALPHA * e;
      e = ad[i] > 0 ? e : MASK_VAL;
      ev[i] = e;
      tm = fmaxf(tm, e);
    }
    // ---- online softmax statistics (8-lane row reduction) ----
    tm = fmaxf(tm, __shfl_xor(tm, 1, 8));
    tm = fmaxf(tm, __shfl_xor(tm, 2, 8));
    tm = fmaxf(tm, __shfl_xor(tm, 4, 8));
    float m_new = fmaxf(m_run, tm);
    float scale = __expf(m_run - m_new);   // exp(-inf)=0 on first tile
    float rs = 0.f;
    v8h pv;
#pragma unroll
    for (int i = 0; i < 8; ++i) {
      float p = __expf(ev[i] - m_new);
      rs += p;
      pv[i] = (_Float16)p;
    }
    rs += __shfl_xor(rs, 1, 8);
    rs += __shfl_xor(rs, 2, 8);
    rs += __shfl_xor(rs, 4, 8);
    l_run = l_run * scale + rs;
    m_run = m_new;

    *(v8h*)&P[b][row][c0] = pv;           // one ds_store_b128
    if (cg == 0) rowf[b][row] = scale;
    __syncthreads();                       // publish P/rowf buffer b

    // ---- rescale accumulators by per-row factor ----
#pragma unroll
    for (int k = 0; k < 8; ++k) {
      float sc = rowf[b][k + 8 * g];
      acc0[k] *= sc;
      acc1[k] *= sc;
    }

    // ---- P (16x64 f16) x WhhT tile (64x16 f16): 2 K-steps x 2 N-tiles ----
#pragma unroll
    for (int kk = 0; kk < 64; kk += 32) {
      V16 Af;
      Af.h[0] = *(const v8h*)&P[b][m][kk + 16 * g];
      Af.h[1] = *(const v8h*)&P[b][m][kk + 16 * g + 8];
      {
        const _Float16* bp = WhhT + (size_t)((2 * w + 0) * 16 + m) * N_NODES + j0 + kk + 16 * g;
        V16 B; B.h[0] = *(const v8h*)bp; B.h[1] = *(const v8h*)(bp + 8);
        acc0 = __builtin_amdgcn_wmma_f32_16x16x32_f16(false, Af.v, false, B.v,
                                                      (short)0, acc0, false, false);
      }
      {
        const _Float16* bp = WhhT + (size_t)((2 * w + 1) * 16 + m) * N_NODES + j0 + kk + 16 * g;
        V16 B; B.h[0] = *(const v8h*)bp; B.h[1] = *(const v8h*)(bp + 8);
        acc1 = __builtin_amdgcn_wmma_f32_16x16x32_f16(false, Af.v, false, B.v,
                                                      (short)0, acc1, false, false);
      }
    }
    // no trailing barrier: next iteration writes the other buffer; the next
    // publish barrier orders any re-write of this buffer after these reads.
  }

  // ---- finalize: divide by softmax denom, ELU, store ----
  __syncthreads();                         // protect last buffer before reuse
  if (cg == 0) rowf[0][row] = 1.0f / l_run;
  __syncthreads();
#pragma unroll
  for (int k = 0; k < 8; ++k) {
    float inv = rowf[0][k + 8 * g];
    float v0 = acc0[k] * inv;
    v0 = v0 > 0.f ? v0 : (__expf(v0) - 1.f);
    out[(size_t)(i0 + k + 8 * g) * OUT_DIM + (2 * w + 0) * 16 + m] = v0;
    float v1 = acc1[k] * inv;
    v1 = v1 > 0.f ? v1 : (__expf(v1) - 1.f);
    out[(size_t)(i0 + k + 8 * g) * OUT_DIM + (2 * w + 1) * 16 + m] = v1;
  }
}

// ---------------------------------------------------------------------------
extern "C" void kernel_launch(void* const* d_in, const int* in_sizes, int n_in,
                              void* d_out, int out_size, void* d_ws, size_t ws_size,
                              hipStream_t stream) {
  const int*   adj = (const int*)d_in[0];
  const float* X   = (const float*)d_in[1];
  const float* W   = (const float*)d_in[2];
  const float* a   = (const float*)d_in[3];
  float* out = (float*)d_out;

  char* ws = (char*)d_ws;
  _Float16* WT    = (_Float16*)(ws);                                  // 64 KB
  _Float16* WhhT  = (_Float16*)(ws + (64 << 10));                     // 2 MB
  float*    s1    = (float*)(ws + (64 << 10) + (2 << 20));            // 32 KB
  float*    s2    = (float*)(ws + (64 << 10) + (2 << 20) + (32 << 10));

  gat_wt_kernel<<<(IN_DIM * OUT_DIM + 255) / 256, 256, 0, stream>>>(W, WT);
  gat_wh_kernel<<<N_NODES / 16, 128, 0, stream>>>(X, WT, a, WhhT, s1, s2);
  gat_attn_kernel<<<N_NODES / 16, 128, 0, stream>>>(adj, WhhT, s1, s2, out);
}
